// CurricularFace_18691697672372
// MI455X (gfx1250) — matrix operand order repaired and checked by reading, not compiled, not run on hardware
//
#include <hip/hip_runtime.h>
#include <math.h>

// CurricularFace constants (m = 0.5, s = 30)
#define S_SCALE   30.0f
#define COS_M_C    0.8775825618903728f   // cos(0.5)
#define SIN_M_C    0.4794255386042030f   // sin(0.5)
#define THRESH_C  -0.8775825618903728f   // cos(pi - 0.5)
#define MM_C       0.2397127693021015f   // sin(pi - 0.5) * 0.5

typedef __bf16 v16bf __attribute__((ext_vector_type(16)));
typedef __bf16 v8bf  __attribute__((ext_vector_type(8)));
typedef float  v8f   __attribute__((ext_vector_type(8)));
typedef unsigned int u32x4 __attribute__((ext_vector_type(4)));
typedef int i32x4 __attribute__((ext_vector_type(4)));
typedef int i32x8 __attribute__((ext_vector_type(8)));

#if defined(__has_builtin)
#if __has_builtin(__builtin_amdgcn_tensor_load_to_lds) && \
    __has_builtin(__builtin_amdgcn_s_wait_tensorcnt)
#define HAS_TDM 1
#endif
#endif

__device__ __forceinline__ unsigned short f32_to_bf16_rne(float f) {
  unsigned u = __float_as_uint(f);
  unsigned r = u + 0x7FFFu + ((u >> 16) & 1u);
  return (unsigned short)(r >> 16);
}

// Low 32 bits of a flat pointer into the LDS aperture == LDS byte address
// (ISA 10.2).
__device__ __forceinline__ unsigned lds_byte_addr(const void* p) {
  return (unsigned)(unsigned long long)(uintptr_t)p;
}

// ---------------------------------------------------------------------------
// Kernel 1: L2-normalize each embedding row; emit f32 + bf16 copies.
// ---------------------------------------------------------------------------
__global__ void __launch_bounds__(256)
rownorm_kernel(const float* __restrict__ emb,
               float* __restrict__ embN,
               unsigned short* __restrict__ embNbf,
               int D) {
  __shared__ float red[256];
  const int row = blockIdx.x, tid = threadIdx.x;
  const float* src = emb + (size_t)row * D;
  float e0 = src[tid];
  float e1 = src[tid + 256];
  red[tid] = e0 * e0 + e1 * e1;
  __syncthreads();
  for (int s = 128; s > 0; s >>= 1) {
    if (tid < s) red[tid] += red[tid + s];
    __syncthreads();
  }
  float inv = rsqrtf(red[0] + 1e-30f);
  float n0 = e0 * inv, n1 = e1 * inv;
  embN[(size_t)row * D + tid]        = n0;
  embN[(size_t)row * D + tid + 256]  = n1;
  embNbf[(size_t)row * D + tid]       = f32_to_bf16_rne(n0);
  embNbf[(size_t)row * D + tid + 256] = f32_to_bf16_rne(n1);
}

// ---------------------------------------------------------------------------
// Kernel 2: per-row target logit via gathered column dot product.
// ---------------------------------------------------------------------------
__global__ void __launch_bounds__(256)
target_kernel(const float* __restrict__ embN,
              const float* __restrict__ kern,
              const int* __restrict__ label,
              float* __restrict__ tl,
              float* __restrict__ ctm,
              float* __restrict__ ft,
              int D, int C) {
  __shared__ float rd[256];
  __shared__ float rq[256];
  const int row = blockIdx.x, tid = threadIdx.x;
  const int col = label[row];
  float dot = 0.f, qq = 0.f;
  for (int d = tid; d < D; d += 256) {
    float kv = kern[(size_t)d * C + col];
    dot += embN[(size_t)row * D + d] * kv;
    qq  += kv * kv;
  }
  rd[tid] = dot; rq[tid] = qq;
  __syncthreads();
  for (int s = 128; s > 0; s >>= 1) {
    if (tid < s) { rd[tid] += rd[tid + s]; rq[tid] += rq[tid + s]; }
    __syncthreads();
  }
  if (tid == 0) {
    float c = rd[0] * rsqrtf(rq[0] + 1e-30f);
    c = fminf(1.f, fmaxf(-1.f, c));
    float st = sqrtf(fmaxf(0.f, 1.f - c * c));
    float cm = c * COS_M_C - st * SIN_M_C;
    tl[row]  = c;
    ctm[row] = cm;
    ft[row]  = (c > THRESH_C) ? cm : (c - MM_C);
  }
}

// ---------------------------------------------------------------------------
// Kernel 3: t_new = 0.01 * mean(tl) + 0.99 * t
// ---------------------------------------------------------------------------
__global__ void __launch_bounds__(512)
tnew_kernel(const float* __restrict__ tl,
            const float* __restrict__ t_in,
            float* __restrict__ tnew, int B) {
  __shared__ float r[512];
  const int tid = threadIdx.x;
  r[tid] = tl[tid];
  __syncthreads();
  for (int s = 256; s > 0; s >>= 1) {
    if (tid < s) r[tid] += r[tid + s];
    __syncthreads();
  }
  if (tid == 0) tnew[0] = (r[0] / (float)B) * 0.01f + 0.99f * t_in[0];
}

// ---------------------------------------------------------------------------
// Kernel 4: fused bf16-WMMA GEMM (M=512 full, N_TILE=32, K chunked by 64).
// A tile (512x64 bf16) staged global->LDS by the Tensor Data Mover with
// per-row padding (128B data + 16B pad -> 144B stride, conflict-free ds).
// B tile staged through VGPRs (needs f32->bf16 cvt + fused column sumsq).
// Kernel matrix read from HBM exactly once; epilogue fused, output *S.
// ---------------------------------------------------------------------------
__global__ void __launch_bounds__(256)
gemm_epilogue_kernel(const unsigned short* __restrict__ embNbf,
                     const float* __restrict__ kern,
                     const int* __restrict__ label,
                     const float* __restrict__ ctm,
                     const float* __restrict__ ft,
                     const float* __restrict__ tnewp,
                     float* __restrict__ out,
                     int C) {
  const int D = 512;
  const int AST = 72;                 // A row stride in shorts (144 B)
  const int BST = 72;                 // B row stride in shorts (144 B)
  __shared__ __align__(16) unsigned short a_lds[512 * 72]; // 72 KB
  __shared__ __align__(16) unsigned short b_lds[32 * 72];  // 4.5 KB
  __shared__ float csq_part[256][4];
  __shared__ float inv_nrm[32];
  __shared__ float s_ctm[512];
  __shared__ float s_ft[512];
  __shared__ int   s_lab[512];

  const int tid  = threadIdx.x;
  const int lane = tid & 31;
  const int wid  = tid >> 5;        // wave id 0..7 -> M group (64 rows)
  const int half = lane >> 4;       // 0 | 1
  const int l16  = lane & 15;
  const int cbase = blockIdx.x * 32;

  for (int i = tid; i < 512; i += 256) {
    s_ctm[i] = ctm[i];
    s_ft[i]  = ft[i];
    s_lab[i] = label[i];
  }
  const float tn = tnewp[0];

  // B staging: thread -> (k = tid/8, 4 consecutive cols q*4..q*4+3)
  const int bq = tid & 7;
  const int bk = tid >> 3;
  int c4 = cbase + bq * 4;
  if (c4 + 3 >= C) c4 = C - 4;

#ifdef HAS_TDM
  // TDM descriptor group 1 (constant across chunks):
  //   data_size=2B, pad_enable, pad_interval=32 DWORDs, pad_amount=4 DWORDs,
  //   tensor_dim0=512, tensor_dim1=512, tile_dim0=64, tile_dim1=512,
  //   tensor_dim0_stride=512, tensor_dim1_stride=512.
  const i32x8 tdm_g1 = {
      (int)((1u << 16) | (1u << 20) | (4u << 22) | (3u << 25)),
      (int)(512u << 16),              // tensor_dim0[15:0]
      (int)(512u << 16),              // tensor_dim1[15:0]
      (int)(64u << 16),               // tile_dim0
      (int)512,                       // tile_dim1 (tile_dim2 = 0)
      (int)512,                       // tensor_dim0_stride[31:0]
      (int)(512u << 16),              // tensor_dim1_stride[15:0]
      0
  };
  const i32x4 z4 = {0, 0, 0, 0};
  const unsigned a_lds_base = lds_byte_addr(&a_lds[0]);
#endif

  float csq[4] = {0.f, 0.f, 0.f, 0.f};
  v8f acc[4][2] = {};

  for (int k0 = 0; k0 < D; k0 += 64) {
#ifdef HAS_TDM
    if (wid == 0) {
      unsigned long long ga =
          (unsigned long long)(uintptr_t)(const void*)(embNbf + k0);
      u32x4 g0;
      g0[0] = 1u;                                 // count=1 (valid descriptor)
      g0[1] = a_lds_base;                         // lds_addr
      g0[2] = (unsigned)ga;                       // global_addr[31:0]
      g0[3] = (unsigned)(ga >> 32) | (2u << 30);  // global_addr[56:32] | type=2
#if __clang_major__ >= 23
      const i32x8 z8 = {0, 0, 0, 0, 0, 0, 0, 0};
      __builtin_amdgcn_tensor_load_to_lds(g0, tdm_g1, z4, z4, z8, 0);
#else
      __builtin_amdgcn_tensor_load_to_lds(g0, tdm_g1, z4, z4, 0);
#endif
    }
#else
    {
      // Fallback: synchronous A staging (2 rows x 128B per thread).
      const int r0 = tid * 2;
      const uint4* s0 = (const uint4*)(embNbf + (size_t)r0 * D + k0);
      uint4* d0 = (uint4*)&a_lds[r0 * AST];
      const uint4* s1 = (const uint4*)(embNbf + (size_t)(r0 + 1) * D + k0);
      uint4* d1 = (uint4*)&a_lds[(r0 + 1) * AST];
#pragma unroll
      for (int j = 0; j < 8; ++j) { d0[j] = s0[j]; d1[j] = s1[j]; }
    }
#endif

    // ---- stage B: one b128 per thread, fused column sumsq, cvt to bf16 ----
    {
      const float4 v = *(const float4*)&kern[(size_t)(k0 + bk) * C + c4];
      csq[0] += v.x * v.x; csq[1] += v.y * v.y;
      csq[2] += v.z * v.z; csq[3] += v.w * v.w;
      b_lds[(bq * 4 + 0) * BST + bk] = f32_to_bf16_rne(v.x);
      b_lds[(bq * 4 + 1) * BST + bk] = f32_to_bf16_rne(v.y);
      b_lds[(bq * 4 + 2) * BST + bk] = f32_to_bf16_rne(v.z);
      b_lds[(bq * 4 + 3) * BST + bk] = f32_to_bf16_rne(v.w);
    }
    if (k0 + 64 < D)  // hint next K chunk (global_prefetch_b8)
      __builtin_prefetch(&kern[(size_t)(k0 + 64 + bk) * C + c4], 0, 1);

#ifdef HAS_TDM
    if (wid == 0) __builtin_amdgcn_s_wait_tensorcnt(0);
#endif
    __syncthreads();

    // ---- two K=32 sub-steps per chunk ----
#pragma unroll
    for (int kk = 0; kk < 64; kk += 32) {
      v16bf bfrag[2];
#pragma unroll
      for (int ni = 0; ni < 2; ++ni) {
        const int base = (ni * 16 + l16) * BST + kk + half * 16;
        v8bf b0 = *(const v8bf*)&b_lds[base];
        v8bf b1 = *(const v8bf*)&b_lds[base + 8];
        v16bf b;
#pragma unroll
        for (int e = 0; e < 8; ++e) { b[e] = b0[e]; b[e + 8] = b1[e]; }
        bfrag[ni] = b;
      }
      v16bf afrag[4];
#pragma unroll
      for (int mi = 0; mi < 4; ++mi) {
        const int rb = (wid * 64 + mi * 16 + l16) * AST + kk + half * 16;
        v8bf s0 = *(const v8bf*)&a_lds[rb];        // K 0-7  / 16-23
        v8bf s1 = *(const v8bf*)&a_lds[rb + 8];    // K 8-15 / 24-31
        v16bf a;
#pragma unroll
        for (int e = 0; e < 8; ++e) { a[e] = s0[e]; a[e + 8] = s1[e]; }
        afrag[mi] = a;
      }
#pragma unroll
      for (int mi = 0; mi < 4; ++mi)
#pragma unroll
        for (int ni = 0; ni < 2; ++ni)
          acc[mi][ni] = __builtin_amdgcn_wmma_f32_16x16x32_bf16(
              false, afrag[mi], false, bfrag[ni],
              (short)0, acc[mi][ni], false, false);
    }
    __syncthreads();
  }

  // ---- column inverse norms (col c <- threads t with t%8 == c/4) ----
  csq_part[tid][0] = csq[0];
  csq_part[tid][1] = csq[1];
  csq_part[tid][2] = csq[2];
  csq_part[tid][3] = csq[3];
  __syncthreads();
  if (tid < 32) {
    const int q = tid >> 2, e = tid & 3;
    float s = 0.f;
#pragma unroll
    for (int k = 0; k < 32; ++k) s += csq_part[k * 8 + q][e];
    inv_nrm[tid] = rsqrtf(s + 1e-30f);
  }
  __syncthreads();

  // ---- fused CurricularFace epilogue; C layout: (vgpr j, lane l) ->
  //      M = j + 8*(l>=16), N = l%16 ----
#pragma unroll
  for (int mi = 0; mi < 4; ++mi) {
#pragma unroll
    for (int ni = 0; ni < 2; ++ni) {
      const int coll = ni * 16 + l16;
      const int gcol = cbase + coll;
      if (gcol < C) {
        const float inv = inv_nrm[coll];
        v8f a = acc[mi][ni];
#pragma unroll
        for (int j = 0; j < 8; ++j) {
          const int grow = wid * 64 + mi * 16 + half * 8 + j;
          float cosine = fminf(1.f, fmaxf(-1.f, a[j] * inv));
          float res;
          if (s_lab[grow] == gcol) {
            res = s_ft[grow];
          } else {
            float cm = s_ctm[grow];
            res = (cosine > cm) ? cosine * (tn + cosine) : cosine;
          }
          out[(size_t)grow * C + gcol] = res * S_SCALE;
        }
      }
    }
  }
}

// ---------------------------------------------------------------------------
extern "C" void kernel_launch(void* const* d_in, const int* in_sizes, int n_in,
                              void* d_out, int out_size, void* d_ws, size_t ws_size,
                              hipStream_t stream) {
  const float* emb   = (const float*)d_in[0];
  const float* kern  = (const float*)d_in[1];
  const int*   label = (const int*)d_in[2];
  const float* t_in  = (const float*)d_in[3];
  float* out = (float*)d_out;

  const int B = in_sizes[2];            // 512
  const int D = in_sizes[0] / B;        // 512
  const int C = in_sizes[1] / D;        // 100000

  char* w = (char*)d_ws;
  unsigned short* embNbf = (unsigned short*)w;                    // B*D bf16
  float* embN = (float*)(w + (size_t)B * D * sizeof(unsigned short));
  float* tl   = (float*)((char*)embN + (size_t)B * D * sizeof(float));
  float* ctm  = tl + B;
  float* ft   = ctm + B;
  float* tnew = ft + B;

  rownorm_kernel<<<B, 256, 0, stream>>>(emb, embN, embNbf, D);
  target_kernel<<<B, 256, 0, stream>>>(embN, kern, label, tl, ctm, ft, D, C);
  tnew_kernel<<<1, 512, 0, stream>>>(tl, t_in, tnew, B);

  const int ntiles = (C + 31) / 32;
  gemm_epilogue_kernel<<<ntiles, 256, 0, stream>>>(embNbf, kern, label, ctm, ft,
                                                   tnew, out, C);
}
